// LRPCHead_661424964067
// MI455X (gfx1250) — matrix operand order repaired and checked by reading, not compiled, not run on hardware
//
#include <hip/hip_runtime.h>
#include <hip/hip_bf16.h>

typedef _Float16 v4h  __attribute__((ext_vector_type(4)));
typedef _Float16 v8h  __attribute__((ext_vector_type(8)));
typedef _Float16 v16h __attribute__((ext_vector_type(16)));
typedef float    v8f  __attribute__((ext_vector_type(8)));

#define CCH 512
#define HWP 25600
#define NCV 1203
#define LCH 64

#define BM 64
#define BN 128
#define BK 32
#define BSTRIDE 40   // halves per B row in LDS (80 bytes, 16B aligned, padded vs 32)

// ---------------- Kernel 1: proposal-filter score -> float mask ----------------
__global__ __launch_bounds__(256) void pf_mask_kernel(
    const float* __restrict__ cls, const float* __restrict__ pf_w,
    const float* __restrict__ pf_b, const float* __restrict__ conf,
    float* __restrict__ maskf)
{
    const int p = blockIdx.x * 256 + threadIdx.x;
    float acc = 0.0f;
#pragma unroll 8
    for (int c = 0; c < CCH; ++c)
        acc = fmaf(pf_w[c], cls[(size_t)c * HWP + p], acc);
    acc += pf_b[0];
    const float cf  = conf[0];
    const float thr = __logf(cf / (1.0f - cf));   // sigmoid(x)>cf  <=>  x>logit(cf)
    maskf[p] = (acc > thr) ? 1.0f : 0.0f;
}

// ---------------- Kernel 2: loc 1x1 conv (64->64), memory-bound ----------------
__global__ __launch_bounds__(256) void loc_kernel(
    const float* __restrict__ loc_feat, const float* __restrict__ loc_w,
    const float* __restrict__ loc_b, float* __restrict__ loc_out)
{
    const int p = blockIdx.x * 256 + threadIdx.x;
    for (int oc = 0; oc < LCH; oc += 16) {
        float acc[16];
#pragma unroll
        for (int o = 0; o < 16; ++o) acc[o] = loc_b[oc + o];
        for (int i = 0; i < LCH; ++i) {
            const float fv = loc_feat[(size_t)i * HWP + p];    // L2-resident after 1st chunk
#pragma unroll
            for (int o = 0; o < 16; ++o)
                acc[o] = fmaf(loc_w[(oc + o) * LCH + i], fv, acc[o]); // uniform -> s_load
        }
#pragma unroll
        for (int o = 0; o < 16; ++o)
            __builtin_nontemporal_store(acc[o], &loc_out[(size_t)(oc + o) * HWP + p]);
    }
}

// ---------------- Kernel 3: vocab GEMM with WMMA f16 (f32 accum) ----------------
// D[v,p] = mask[p] * ( sum_k vocab_w[v,k] * cls_feat[k,p] ) + vocab_b[v]
__global__ __launch_bounds__(256) void vocab_gemm_kernel(
    const float* __restrict__ A,      // vocab_w [NCV, CCH] row-major
    const float* __restrict__ Bg,     // cls_feat [CCH, HWP] row-major
    const float* __restrict__ bias,   // vocab_b [NCV]
    const float* __restrict__ maskf,  // [HWP] 0/1
    float* __restrict__ D)            // cls_out [NCV, HWP]
{
    __shared__ _Float16 Als[BM * BK];        // [m][k], row stride 32 halves (64B)
    __shared__ _Float16 Bls[BN * BSTRIDE];   // [n][k], row stride 40 halves (80B)

    const int tid  = threadIdx.x;
    const int lane = tid & 31;
    const int wave = tid >> 5;         // 8 waves
    const int wm   = wave >> 2;        // 0..1 : 32-row strip
    const int wn   = wave & 3;         // 0..3 : 32-col strip
    const int hi   = lane >> 4;        // lane half (WMMA layout)
    const int l16  = lane & 15;

    const int nBase = blockIdx.x * BN;
    const int mBase = blockIdx.y * BM;

    // A loader: thread covers (m = tid/8 [+32], k4 = (tid%8)*4).
    // Rows >= NCV are CLAMPED (not zeroed): their accumulators are never
    // written by the row-guarded epilogue, so garbage there is harmless and
    // we avoid exec-masked branches in the hot loop.
    const int am = tid >> 3;
    const int ak = (tid & 7) * 4;
    const int aRow0 = min(mBase + am,      NCV - 1);
    const int aRow1 = min(mBase + am + 32, NCV - 1);
    // B loader: thread covers (k = wave [+8,+16,+24], n4 = lane*4)
    const int bk  = wave;
    const int bn4 = lane * 4;

    v8f acc[2][2] = {};

    // -------- software pipeline: prefetch K-step 0 into registers --------
    float4 aReg[2], bReg[4];
    aReg[0] = *(const float4*)(A + (size_t)aRow0 * CCH + ak);
    aReg[1] = *(const float4*)(A + (size_t)aRow1 * CCH + ak);
#pragma unroll
    for (int kk = 0; kk < 4; ++kk)
        bReg[kk] = *(const float4*)(Bg + (size_t)(bk + kk * 8) * HWP + nBase + bn4);

    for (int ks = 0; ks < CCH; ks += BK) {
        // ---- stage current tiles into LDS (f32 -> f16 convert) ----
#pragma unroll
        for (int mm = 0; mm < 2; ++mm) {
            v4h ah;
            ah[0] = (_Float16)aReg[mm].x; ah[1] = (_Float16)aReg[mm].y;
            ah[2] = (_Float16)aReg[mm].z; ah[3] = (_Float16)aReg[mm].w;
            *(v4h*)(&Als[(am + mm * 32) * BK + ak]) = ah;
        }
#pragma unroll
        for (int kk = 0; kk < 4; ++kk) {
            const int krow = bk + kk * 8;
            Bls[(bn4 + 0) * BSTRIDE + krow] = (_Float16)bReg[kk].x;
            Bls[(bn4 + 1) * BSTRIDE + krow] = (_Float16)bReg[kk].y;
            Bls[(bn4 + 2) * BSTRIDE + krow] = (_Float16)bReg[kk].z;
            Bls[(bn4 + 3) * BSTRIDE + krow] = (_Float16)bReg[kk].w;
        }
        __syncthreads();

        // ---- prefetch next K-step: latency hidden under frag loads + WMMA ----
        const int kn = ks + BK;
        if (kn < CCH) {
            aReg[0] = *(const float4*)(A + (size_t)aRow0 * CCH + kn + ak);
            aReg[1] = *(const float4*)(A + (size_t)aRow1 * CCH + kn + ak);
#pragma unroll
            for (int kk = 0; kk < 4; ++kk)
                bReg[kk] = *(const float4*)(Bg + (size_t)(kn + bk + kk * 8) * HWP +
                                            nBase + bn4);
        }

        // ---- fragments per ISA 16-bit layouts ----
        // A 16x32: lane row = l16; halves {hi*8..hi*8+7} and {16+hi*8..}
        v16h aF[2], bF[2];
#pragma unroll
        for (int i = 0; i < 2; ++i) {
            const _Float16* base = &Als[(wm * 32 + i * 16 + l16) * BK + hi * 8];
            const v8h lo = *(const v8h*)(base);
            const v8h hh = *(const v8h*)(base + 16);
#pragma unroll
            for (int e = 0; e < 8; ++e) { aF[i][e] = lo[e]; aF[i][8 + e] = hh[e]; }
        }
        // B 32x16: lane col = l16; halves {hi*8..hi*8+7} and {16+hi*8..}
#pragma unroll
        for (int j = 0; j < 2; ++j) {
            const _Float16* base = &Bls[(wn * 32 + j * 16 + l16) * BSTRIDE + hi * 8];
            const v8h lo = *(const v8h*)(base);
            const v8h hh = *(const v8h*)(base + 16);
#pragma unroll
            for (int e = 0; e < 8; ++e) { bF[j][e] = lo[e]; bF[j][8 + e] = hh[e]; }
        }

#pragma unroll
        for (int i = 0; i < 2; ++i)
#pragma unroll
            for (int j = 0; j < 2; ++j)
                acc[i][j] = __builtin_amdgcn_wmma_f32_16x16x32_f16(
                    false, aF[i], false, bF[j], (short)0, acc[i][j], false, false);
        __syncthreads();
    }

    // ---- epilogue: out = mask*acc + bias; NT stores keep cls_feat in L2 ----
#pragma unroll
    for (int j = 0; j < 2; ++j) {
        const int pcol = nBase + wn * 32 + j * 16 + l16;
        const float mv = maskf[pcol];
#pragma unroll
        for (int i = 0; i < 2; ++i) {
#pragma unroll
            for (int e = 0; e < 8; ++e) {
                const int vrow = mBase + wm * 32 + i * 16 + hi * 8 + e;
                if (vrow < NCV)
                    __builtin_nontemporal_store(fmaf(mv, acc[i][j][e], bias[vrow]),
                                                &D[(size_t)vrow * HWP + pcol]);
            }
        }
    }
}

extern "C" void kernel_launch(void* const* d_in, const int* in_sizes, int n_in,
                              void* d_out, int out_size, void* d_ws, size_t ws_size,
                              hipStream_t stream) {
    const float* cls_feat = (const float*)d_in[0];
    const float* loc_feat = (const float*)d_in[1];
    const float* conf     = (const float*)d_in[2];
    const float* pf_w     = (const float*)d_in[3];
    const float* pf_b     = (const float*)d_in[4];
    const float* vocab_w  = (const float*)d_in[5];
    const float* vocab_b  = (const float*)d_in[6];
    const float* loc_w    = (const float*)d_in[7];
    const float* loc_b    = (const float*)d_in[8];

    float* loc_out = (float*)d_out;                       // [64, 25600]
    float* cls_out = loc_out + (size_t)LCH * HWP;         // [1203, 25600]
    float* maskf   = cls_out + (size_t)NCV * HWP;         // [25600] as 0/1 float

    // 1) mask (also warms L2 with cls_feat; 52MB fits in 192MB L2)
    pf_mask_kernel<<<HWP / 256, 256, 0, stream>>>(cls_feat, pf_w, pf_b, conf, maskf);
    // 2) loc 1x1 conv (independent, memory-bound)
    loc_kernel<<<HWP / 256, 256, 0, stream>>>(loc_feat, loc_w, loc_b, loc_out);
    // 3) vocab GEMM, mask applied in epilogue (same-stream order guarantees mask ready)
    dim3 grid(HWP / BN, (NCV + BM - 1) / BM);   // 200 x 19
    vocab_gemm_kernel<<<grid, 256, 0, stream>>>(vocab_w, cls_feat, vocab_b, maskf, cls_out);
}
